// Decoder_33200097198882
// MI455X (gfx1250) — matrix operand order repaired and checked by reading, not compiled, not run on hardware
//
#include <hip/hip_runtime.h>
#include <hip/hip_bf16.h>
#include <math.h>

// Problem constants (fixed by the reference).
#define BB   64
#define LL   1024
#define HH   512
#define AA   1024     // attention dim == 2H == K of both big GEMMs
#define EE   256
#define II   256
#define VV   50000
#define OOV  100
#define WOUT (VV + OOV)   // 50100

typedef __attribute__((ext_vector_type(16))) __bf16 v16bf;
typedef __attribute__((ext_vector_type(4)))  __bf16 v4bf;
typedef __attribute__((ext_vector_type(8)))  float  v8f;

__device__ __forceinline__ float sigmoidf_(float x) { return 1.0f / (1.0f + __expf(-x)); }

// Branchless tanh: no EXEC divergence, lowers to v_exp/v_rcp/v_cndmask.
__device__ __forceinline__ float tanh_fast(float x) {
  const float ax = fabsf(x);
  const float t  = __expf(-2.0f * ax);
  const float r  = (1.0f - t) / (1.0f + t);
  return copysignf(r, x);
}

__device__ __forceinline__ v4bf cvt4(const float4 f) {
  v4bf p;
  p[0] = (__bf16)f.x; p[1] = (__bf16)f.y; p[2] = (__bf16)f.z; p[3] = (__bf16)f.w;
  return p;
}

// ---------------------------------------------------------------------------
// K1: LSTM step with h0=c0=0  ->  state_cell = [h | c]  (64 x 1024)
// ---------------------------------------------------------------------------
__global__ void k_lstm(const float* __restrict__ y, const float* __restrict__ Wih,
                       const float* __restrict__ bih, const float* __restrict__ bhh,
                       float* __restrict__ sc) {
  __shared__ float sy[II];
  const int b = blockIdx.x;
  const int j = blockIdx.y * 256 + threadIdx.x;   // 0..511
  sy[threadIdx.x] = y[b * II + threadIdx.x];
  __syncthreads();
  const float4* ri = (const float4*)(Wih + (size_t)j * II);
  const float4* rg = (const float4*)(Wih + (size_t)(2 * HH + j) * II);
  const float4* ro = (const float4*)(Wih + (size_t)(3 * HH + j) * II);
  float gi = 0.f, gg = 0.f, go = 0.f;
  #pragma unroll 4
  for (int k = 0; k < II / 4; ++k) {
    const float4 xv = ((const float4*)sy)[k];
    const float4 a = ri[k], g = rg[k], o = ro[k];
    gi += xv.x * a.x + xv.y * a.y + xv.z * a.z + xv.w * a.w;
    gg += xv.x * g.x + xv.y * g.y + xv.z * g.z + xv.w * g.w;
    go += xv.x * o.x + xv.y * o.y + xv.z * o.z + xv.w * o.w;
  }
  gi += bih[j] + bhh[j];
  gg += bih[2 * HH + j] + bhh[2 * HH + j];
  go += bih[3 * HH + j] + bhh[3 * HH + j];
  const float c = sigmoidf_(gi) * tanh_fast(gg);   // f-gate * c0 == 0
  const float h = sigmoidf_(go) * tanh_fast(c);
  sc[b * AA + j] = h;
  sc[b * AA + HH + j] = c;
}

// ---------------------------------------------------------------------------
// K2: bias[b,n] = state_cell[b,:] . Ws_w[n,:] + Ws_b[n] + Wh_b[n]
// ---------------------------------------------------------------------------
__global__ void k_bias(const float* __restrict__ sc, const float* __restrict__ Wsw,
                       const float* __restrict__ Wsb, const float* __restrict__ Whb,
                       float* __restrict__ bias) {
  __shared__ float ss[AA];
  const int b = blockIdx.x;
  const int n = blockIdx.y * 256 + threadIdx.x;
  for (int i = threadIdx.x; i < AA; i += 256) ss[i] = sc[b * AA + i];
  __syncthreads();
  const float4* row = (const float4*)(Wsw + (size_t)n * AA);
  float acc = 0.f;
  #pragma unroll 4
  for (int k = 0; k < AA / 4; ++k) {
    const float4 s4 = ((const float4*)ss)[k];
    const float4 w4 = row[k];
    acc += s4.x * w4.x + s4.y * w4.y + s4.z * w4.z + s4.w * w4.w;
  }
  bias[b * AA + n] = acc + Wsb[n] + Whb[n];
}

// ---------------------------------------------------------------------------
// K3: fused attention scores via bf16 WMMA, double-buffered LDS.
// scores[b,l] = sum_n v[n] * tanh( enc[b,l,:] . Wh_w[n,:] + bias[b,n] )
// Block: 256 thr (8 waves), M-tile = 64 l-rows of one batch, N looped in 128s.
// ---------------------------------------------------------------------------
__global__ void k_scores(const float* __restrict__ enc, const float* __restrict__ Whw,
                         const float* __restrict__ bias, const float* __restrict__ v,
                         float* __restrict__ scores) {
  __shared__ __bf16 sA[2][64][40];     // ping-pong 64 x 32K tiles (padded)
  __shared__ __bf16 sB[2][128][40];    // ping-pong 128 x 32K tiles
  __shared__ float  sScore[64];

  const int tid  = threadIdx.x;
  const int b    = blockIdx.x >> 4;
  const int l0   = (blockIdx.x & 15) << 6;
  const int wave = tid >> 5, lane = tid & 31;
  const int mw   = wave & 3, nw = wave >> 2;
  const int half = lane >> 4, ln = lane & 15;

  if (tid < 64) sScore[tid] = 0.0f;

  // per-thread staging coordinates (float4 granularity)
  int aR[2], aC[2], bR[4], bC[4];
  #pragma unroll
  for (int i = 0; i < 2; ++i) { const int q = tid + i * 256; aR[i] = q >> 3; aC[i] = (q & 7) * 4; }
  #pragma unroll
  for (int i = 0; i < 4; ++i) { const int q = tid + i * 256; bR[i] = q >> 3; bC[i] = (q & 7) * 4; }

  float part[8];
  #pragma unroll
  for (int r = 0; r < 8; ++r) part[r] = 0.0f;

  const size_t encBase = ((size_t)b * LL + l0) * AA;
  const int arow = mw * 16 + ln;
  const int koff = half * 8;
  const int kh   = half * 16;

  for (int n0 = 0; n0 < AA; n0 += 128) {
    v8f acc[4];
    #pragma unroll
    for (int s = 0; s < 4; ++s)
      #pragma unroll
      for (int r = 0; r < 8; ++r) acc[s][r] = 0.0f;

    // prologue: stage chunk 0 into buffer 0
    {
      float4 ra[2], rb[4];
      #pragma unroll
      for (int i = 0; i < 2; ++i) ra[i] = *(const float4*)(enc + encBase + (size_t)aR[i] * AA + aC[i]);
      #pragma unroll
      for (int i = 0; i < 4; ++i) rb[i] = *(const float4*)(Whw + (size_t)(n0 + bR[i]) * AA + bC[i]);
      #pragma unroll
      for (int i = 0; i < 2; ++i) *(v4bf*)&sA[0][aR[i]][aC[i]] = cvt4(ra[i]);
      #pragma unroll
      for (int i = 0; i < 4; ++i) *(v4bf*)&sB[0][bR[i]][bC[i]] = cvt4(rb[i]);
    }
    __syncthreads();

    for (int kc = 0; kc < 32; ++kc) {
      const int cur = kc & 1;
      const bool hasNext = (kc + 1) < 32;
      float4 ra[2], rb[4];
      if (hasNext) {
        const int k0 = (kc + 1) * 32;
        #pragma unroll
        for (int i = 0; i < 2; ++i) ra[i] = *(const float4*)(enc + encBase + (size_t)aR[i] * AA + k0 + aC[i]);
        #pragma unroll
        for (int i = 0; i < 4; ++i) rb[i] = *(const float4*)(Whw + (size_t)(n0 + bR[i]) * AA + k0 + bC[i]);
      }
      // compute on current buffer while the loads above are in flight
      v16bf af;
      #pragma unroll
      for (int i = 0; i < 8; ++i) {
        af[i]     = sA[cur][arow][koff + i];
        af[8 + i] = sA[cur][arow][16 + koff + i];
      }
      v16bf bf[4];
      #pragma unroll
      for (int s = 0; s < 4; ++s) {
        const int brow = nw * 64 + s * 16 + ln;
        #pragma unroll
        for (int i = 0; i < 8; ++i) { bf[s][i] = sB[cur][brow][koff + i]; bf[s][8 + i] = sB[cur][brow][16 + koff + i]; }
      }
      #pragma unroll
      for (int s = 0; s < 4; ++s)
        acc[s] = __builtin_amdgcn_wmma_f32_16x16x32_bf16(
            false, af, false, bf[s], (short)0, acc[s], false, false);

      if (hasNext) {
        const int nb = cur ^ 1;
        #pragma unroll
        for (int i = 0; i < 2; ++i) *(v4bf*)&sA[nb][aR[i]][aC[i]] = cvt4(ra[i]);
        #pragma unroll
        for (int i = 0; i < 4; ++i) *(v4bf*)&sB[nb][bR[i]][bC[i]] = cvt4(rb[i]);
      }
      __syncthreads();
    }

    // epilogue: tanh + dot with v over this 128-wide N chunk (branchless)
    #pragma unroll
    for (int s = 0; s < 4; ++s) {
      const int n = n0 + nw * 64 + s * 16 + ln;
      const float vn = v[n];
      const float bn = bias[b * AA + n];
      #pragma unroll
      for (int r = 0; r < 8; ++r) part[r] += vn * tanh_fast(acc[s][r] + bn);
    }
  }
  #pragma unroll
  for (int r = 0; r < 8; ++r)
    atomicAdd(&sScore[mw * 16 + half * 8 + r], part[r]);
  __syncthreads();
  if (tid < 64) scores[b * LL + l0 + tid] = sScore[tid];
}

// ---------------------------------------------------------------------------
// K4: softmax over L per batch -> attn
// ---------------------------------------------------------------------------
__global__ void k_softmax_attn(const float* __restrict__ scores, float* __restrict__ attn) {
  __shared__ float red[256];
  const int b = blockIdx.x, tid = threadIdx.x;
  const float* s = scores + b * LL;
  float m = -1e30f;
  for (int i = tid; i < LL; i += 256) m = fmaxf(m, s[i]);
  red[tid] = m; __syncthreads();
  for (int off = 128; off; off >>= 1) { if (tid < off) red[tid] = fmaxf(red[tid], red[tid + off]); __syncthreads(); }
  m = red[0]; __syncthreads();
  float sum = 0.f;
  for (int i = tid; i < LL; i += 256) sum += __expf(s[i] - m);
  red[tid] = sum; __syncthreads();
  for (int off = 128; off; off >>= 1) { if (tid < off) red[tid] += red[tid + off]; __syncthreads(); }
  const float inv = 1.0f / red[0];
  for (int i = tid; i < LL; i += 256) attn[b * LL + i] = __expf(s[i] - m) * inv;
}

// ---------------------------------------------------------------------------
// K5: ctx[b,d] = sum_l attn[b,l]*enc[b,l,d]  (exact f32, float4 streaming)
// ---------------------------------------------------------------------------
__global__ void k_ctx(const float* __restrict__ attn, const float* __restrict__ enc,
                      float* __restrict__ ctx) {
  __shared__ float sa[LL];
  const int b = blockIdx.x;
  const int d4 = threadIdx.x;                 // float4 column index 0..255
  for (int i = threadIdx.x; i < LL; i += 256) sa[i] = attn[b * LL + i];
  __syncthreads();
  const float4* e = (const float4*)(enc + (size_t)b * LL * AA) + d4;
  float4 a0 = {0,0,0,0}, a1 = {0,0,0,0};
  for (int l = 0; l < LL; l += 2) {
    const float4 t0 = e[(size_t)l * (AA / 4)];
    const float4 t1 = e[(size_t)(l + 1) * (AA / 4)];
    const float w0 = sa[l], w1 = sa[l + 1];
    a0.x += w0 * t0.x; a0.y += w0 * t0.y; a0.z += w0 * t0.z; a0.w += w0 * t0.w;
    a1.x += w1 * t1.x; a1.y += w1 * t1.y; a1.z += w1 * t1.z; a1.w += w1 * t1.w;
  }
  float4 r;
  r.x = a0.x + a1.x; r.y = a0.y + a1.y; r.z = a0.z + a1.z; r.w = a0.w + a1.w;
  ((float4*)ctx)[b * (AA / 4) + d4] = r;
}

// ---------------------------------------------------------------------------
// K6: fc1[b,n] = [ctx | h] . fc1_w[n,:] + fc1_b[n]
// ---------------------------------------------------------------------------
__global__ void k_fc1(const float* __restrict__ ctx, const float* __restrict__ sc,
                      const float* __restrict__ W, const float* __restrict__ bvec,
                      float* __restrict__ out) {
  __shared__ float sin[3 * HH];
  const int b = blockIdx.x;
  const int n = blockIdx.y * 256 + threadIdx.x;
  for (int i = threadIdx.x; i < AA; i += 256) sin[i] = ctx[b * AA + i];
  for (int i = threadIdx.x; i < HH; i += 256) sin[AA + i] = sc[b * AA + i];   // h half
  __syncthreads();
  const float4* row = (const float4*)(W + (size_t)n * (3 * HH));
  float acc = 0.f;
  #pragma unroll 4
  for (int k = 0; k < (3 * HH) / 4; ++k) {
    const float4 s4 = ((const float4*)sin)[k];
    const float4 w4 = row[k];
    acc += s4.x * w4.x + s4.y * w4.y + s4.z * w4.z + s4.w * w4.w;
  }
  out[b * AA + n] = acc + bvec[n];
}

// ---------------------------------------------------------------------------
// K7: p_gen[b] = sigmoid([ctx | state_cell | x] . pgen_w)
// ---------------------------------------------------------------------------
__global__ void k_pgen(const float* __restrict__ ctx, const float* __restrict__ sc,
                       const float* __restrict__ x, const float* __restrict__ w,
                       float* __restrict__ pgen) {
  __shared__ float red[256];
  const int b = blockIdx.x, tid = threadIdx.x;
  float acc = 0.f;
  for (int i = tid; i < AA; i += 256) acc += ctx[b * AA + i] * w[i];
  for (int i = tid; i < AA; i += 256) acc += sc[b * AA + i] * w[AA + i];
  for (int i = tid; i < EE; i += 256) acc += x[b * EE + i] * w[2 * AA + i];
  red[tid] = acc; __syncthreads();
  for (int off = 128; off; off >>= 1) { if (tid < off) red[tid] += red[tid + off]; __syncthreads(); }
  if (tid == 0) pgen[b] = sigmoidf_(red[0]);
}

// ---------------------------------------------------------------------------
// K8: logits (64 x 50000) = fc1 @ fc2_w^T + fc2_b via bf16 WMMA,
// double-buffered; fc2_w streamed from HBM exactly once, converted in LDS.
// ---------------------------------------------------------------------------
__global__ void k_logits(const float* __restrict__ fc1, const float* __restrict__ W2,
                         const float* __restrict__ b2, float* __restrict__ logits) {
  __shared__ __bf16 sA[2][64][40];
  __shared__ __bf16 sB[2][128][40];
  const int tid  = threadIdx.x;
  const int n0   = blockIdx.x * 128;
  const int wave = tid >> 5, lane = tid & 31;
  const int mw   = wave & 3, nw = wave >> 2;
  const int half = lane >> 4, ln = lane & 15;

  int aR[2], aC[2], bR[4], bC[4];
  size_t bOff[4];
  bool bOK[4];
  #pragma unroll
  for (int i = 0; i < 2; ++i) { const int q = tid + i * 256; aR[i] = q >> 3; aC[i] = (q & 7) * 4; }
  #pragma unroll
  for (int i = 0; i < 4; ++i) {
    const int q = tid + i * 256; bR[i] = q >> 3; bC[i] = (q & 7) * 4;
    const int n = n0 + bR[i];
    bOK[i]  = (n < VV);
    bOff[i] = (size_t)(bOK[i] ? n : (VV - 1)) * AA + bC[i];   // clamped, always valid
  }

  v8f acc[4];
  #pragma unroll
  for (int s = 0; s < 4; ++s)
    #pragma unroll
    for (int r = 0; r < 8; ++r) acc[s][r] = 0.0f;

  const float4 f4z = {0, 0, 0, 0};
  // prologue: stage chunk 0 into buffer 0
  {
    float4 ra[2], rb[4];
    #pragma unroll
    for (int i = 0; i < 2; ++i) ra[i] = *(const float4*)(fc1 + (size_t)aR[i] * AA + aC[i]);
    #pragma unroll
    for (int i = 0; i < 4; ++i) rb[i] = bOK[i] ? *(const float4*)(W2 + bOff[i]) : f4z;
    #pragma unroll
    for (int i = 0; i < 2; ++i) *(v4bf*)&sA[0][aR[i]][aC[i]] = cvt4(ra[i]);
    #pragma unroll
    for (int i = 0; i < 4; ++i) *(v4bf*)&sB[0][bR[i]][bC[i]] = cvt4(rb[i]);
  }
  __syncthreads();

  const int arow = mw * 16 + ln;
  const int koff = half * 8;
  const int kh   = half * 16;

  for (int kc = 0; kc < 32; ++kc) {
    const int cur = kc & 1;
    const bool hasNext = (kc + 1) < 32;
    float4 ra[2], rb[4];
    if (hasNext) {
      const int k0 = (kc + 1) * 32;
      #pragma unroll
      for (int i = 0; i < 2; ++i) ra[i] = *(const float4*)(fc1 + (size_t)aR[i] * AA + k0 + aC[i]);
      #pragma unroll
      for (int i = 0; i < 4; ++i) rb[i] = bOK[i] ? *(const float4*)(W2 + bOff[i] + k0) : f4z;
    }
    v16bf af;
    #pragma unroll
    for (int i = 0; i < 8; ++i) {
      af[i]     = sA[cur][arow][koff + i];
      af[8 + i] = sA[cur][arow][16 + koff + i];
    }
    v16bf bf[4];
    #pragma unroll
    for (int s = 0; s < 4; ++s) {
      const int brow = nw * 64 + s * 16 + ln;
      #pragma unroll
      for (int i = 0; i < 8; ++i) { bf[s][i] = sB[cur][brow][koff + i]; bf[s][8 + i] = sB[cur][brow][16 + koff + i]; }
    }
    #pragma unroll
    for (int s = 0; s < 4; ++s)
      acc[s] = __builtin_amdgcn_wmma_f32_16x16x32_bf16(
          false, af, false, bf[s], (short)0, acc[s], false, false);

    if (hasNext) {
      const int nb = cur ^ 1;
      #pragma unroll
      for (int i = 0; i < 2; ++i) *(v4bf*)&sA[nb][aR[i]][aC[i]] = cvt4(ra[i]);
      #pragma unroll
      for (int i = 0; i < 4; ++i) *(v4bf*)&sB[nb][bR[i]][bC[i]] = cvt4(rb[i]);
    }
    __syncthreads();
  }

  #pragma unroll
  for (int s = 0; s < 4; ++s) {
    const int n = n0 + nw * 64 + s * 16 + ln;
    if (n < VV) {
      const float bn = b2[n];
      #pragma unroll
      for (int r = 0; r < 8; ++r) {
        const int m = mw * 16 + half * 8 + r;
        logits[(size_t)m * VV + n] = acc[s][r] + bn;
      }
    }
  }
}

// ---------------------------------------------------------------------------
// K9: per-row max & sum(exp) over 50000 logits
// ---------------------------------------------------------------------------
__global__ void k_rowstats(const float* __restrict__ logits,
                           float* __restrict__ rmax, float* __restrict__ rsum) {
  __shared__ float red[256];
  const int b = blockIdx.x, tid = threadIdx.x;
  const float* row = logits + (size_t)b * VV;
  float m = -1e30f;
  for (int i = tid; i < VV; i += 256) m = fmaxf(m, row[i]);
  red[tid] = m; __syncthreads();
  for (int off = 128; off; off >>= 1) { if (tid < off) red[tid] = fmaxf(red[tid], red[tid + off]); __syncthreads(); }
  m = red[0]; __syncthreads();
  float s = 0.f;
  for (int i = tid; i < VV; i += 256) s += __expf(row[i] - m);
  red[tid] = s; __syncthreads();
  for (int off = 128; off; off >>= 1) { if (tid < off) red[tid] += red[tid + off]; __syncthreads(); }
  if (tid == 0) { rmax[b] = m; rsum[b] = red[0]; }
}

// ---------------------------------------------------------------------------
// K10: p_ext init = p_gen * softmax(logits), zeros over the OOV tail
// ---------------------------------------------------------------------------
__global__ void k_out_init(const float* __restrict__ logits, const float* __restrict__ rmax,
                           const float* __restrict__ rsum, const float* __restrict__ pgen,
                           float* __restrict__ out) {
  const int vv = blockIdx.x * 256 + threadIdx.x;
  const int b  = blockIdx.y;
  if (vv >= WOUT) return;
  float val = 0.0f;
  if (vv < VV)
    val = pgen[b] * __expf(logits[(size_t)b * VV + vv] - rmax[b]) / rsum[b];
  out[(size_t)b * WOUT + vv] = val;
}

// ---------------------------------------------------------------------------
// K11: copy-mechanism scatter: out[b, ids[b,l]] += (1-p_gen[b]) * attn[b,l]
// ---------------------------------------------------------------------------
__global__ void k_scatter(const int* __restrict__ ids, const float* __restrict__ attn,
                          const float* __restrict__ pgen, float* __restrict__ out) {
  const int idx = blockIdx.x * 256 + threadIdx.x;   // 0 .. 64*1024-1
  const int b = idx >> 10;
  const int id = ids[idx];
  const float val = (1.0f - pgen[b]) * attn[idx];
  atomicAdd(&out[(size_t)b * WOUT + id], val);
}

// ---------------------------------------------------------------------------
extern "C" void kernel_launch(void* const* d_in, const int* in_sizes, int n_in,
                              void* d_out, int out_size, void* d_ws, size_t ws_size,
                              hipStream_t stream) {
  const float* x    = (const float*)d_in[0];
  const float* y    = (const float*)d_in[1];
  const float* enc  = (const float*)d_in[2];
  const float* Wih  = (const float*)d_in[3];
  // d_in[4] = W_hh  : unused (h0 == 0)
  const float* bih  = (const float*)d_in[5];
  const float* bhh  = (const float*)d_in[6];
  const float* Wsw  = (const float*)d_in[7];
  const float* Wsb  = (const float*)d_in[8];
  const float* Whw  = (const float*)d_in[9];
  const float* Whb  = (const float*)d_in[10];
  // d_in[11] = wc_w : unused (coverage == 0)
  const float* vw   = (const float*)d_in[12];
  const float* fc1w = (const float*)d_in[13];
  const float* fc1b = (const float*)d_in[14];
  const float* fc2w = (const float*)d_in[15];
  const float* fc2b = (const float*)d_in[16];
  const float* pgw  = (const float*)d_in[17];
  const int*   ids  = (const int*)d_in[18];

  char* ws = (char*)d_ws;
  float* sc     = (float*)ws; ws += (size_t)BB * AA * 4;    // state_cell [h|c]
  float* bias   = (float*)ws; ws += (size_t)BB * AA * 4;    // dec_feat + biases
  float* scores = (float*)ws; ws += (size_t)BB * LL * 4;
  float* attn   = (float*)ws; ws += (size_t)BB * LL * 4;
  float* ctx    = (float*)ws; ws += (size_t)BB * AA * 4;
  float* fc1o   = (float*)ws; ws += (size_t)BB * AA * 4;
  float* pgen   = (float*)ws; ws += 256;
  float* rmax   = (float*)ws; ws += 256;
  float* rsum   = (float*)ws; ws += 256;
  float* logits = (float*)ws; ws += (size_t)BB * VV * 4;

  k_lstm<<<dim3(BB, 2), 256, 0, stream>>>(y, Wih, bih, bhh, sc);
  k_bias<<<dim3(BB, 4), 256, 0, stream>>>(sc, Wsw, Wsb, Whb, bias);
  k_scores<<<BB * (LL / 64), 256, 0, stream>>>(enc, Whw, bias, vw, scores);
  k_softmax_attn<<<BB, 256, 0, stream>>>(scores, attn);
  k_ctx<<<BB, 256, 0, stream>>>(attn, enc, ctx);
  k_fc1<<<dim3(BB, 4), 256, 0, stream>>>(ctx, sc, fc1w, fc1b, fc1o);
  k_pgen<<<BB, 256, 0, stream>>>(ctx, sc, x, pgw, pgen);
  k_logits<<<(VV + 127) / 128, 256, 0, stream>>>(fc1o, fc2w, fc2b, logits);
  k_rowstats<<<BB, 256, 0, stream>>>(logits, rmax, rsum);
  k_out_init<<<dim3((WOUT + 255) / 256, BB), 256, 0, stream>>>(logits, rmax, rsum, pgen, (float*)d_out);
  k_scatter<<<(BB * LL) / 256, 256, 0, stream>>>(ids, attn, pgen, (float*)d_out);
}
